// _PyramidSelfAttentionBlock_85615878079108
// MI455X (gfx1250) — matrix operand order repaired and checked by reading, not compiled, not run on hardware
//
#include <hip/hip_runtime.h>

typedef _Float16 half_t;
typedef half_t v16h __attribute__((ext_vector_type(16)));
typedef half_t v8h  __attribute__((ext_vector_type(8)));
typedef float  v8f  __attribute__((ext_vector_type(8)));

// ---- WMMA helpers (CDNA5 16x16x32 f16 -> f32) -------------------------------

__device__ __forceinline__ v8f wmma_f16(v16h a, v16h b, v8f c) {
  // (neg_a, A, neg_b, B, c_mod, C, reuse_a, reuse_b)
  return __builtin_amdgcn_wmma_f32_16x16x32_f16(false, a, false, b, (short)0, c,
                                                false, false);
}

// A-operand loader: lane holds matrix row `row`; K chunks at [8g,8g+8) and
// [16+8g,16+8g+8)  (16-bit A layout, ISA 7.12.2)
__device__ __forceinline__ v16h ld_a(const half_t* row, int g) {
  v8h lo = *(const v8h*)(row + 8 * g);
  v8h hi = *(const v8h*)(row + 16 + 8 * g);
  v16h r;
#pragma unroll
  for (int j = 0; j < 8; ++j) { r[j] = lo[j]; r[j + 8] = hi[j]; }
  return r;
}

// B-operand loader: lane holds matrix column == LDS row `row`; K chunks at [8g,8g+8) and [16+8g,16+8g+8)
__device__ __forceinline__ v16h ld_b(const half_t* row, int g) {
  v8h lo = *(const v8h*)(row + 8 * g);
  v8h hi = *(const v8h*)(row + 16 + 8 * g);
  v16h r;
#pragma unroll
  for (int j = 0; j < 8; ++j) { r[j] = lo[j]; r[j + 8] = hi[j]; }
  return r;
}

__device__ __forceinline__ float rmax16(float x) {
#pragma unroll
  for (int m = 1; m < 16; m <<= 1) x = fmaxf(x, __shfl_xor(x, m, 32));
  return x;
}
__device__ __forceinline__ float rsum16(float x) {
#pragma unroll
  for (int m = 1; m < 16; m <<= 1) x += __shfl_xor(x, m, 32);
  return x;
}

// ---- fused windowed self-attention ------------------------------------------
// x: [1,64,64,64,64] f32. 512 windows (8x8x8 coarse blocks) of 512 tokens.
// one workgroup (512 thr = 16 waves, wave32) per window.

#define XS_STRIDE 72   // 64 ch + pad  (144B, mult of 16)
#define KS_STRIDE 40   // 32 ch + pad  (80B)
#define VT_STRIDE 520  // 512 tok + pad (1040B)
#define WS_STRIDE 72

#define NWAVE 16
// union region: phase 1/2 -> Xs (512*72 f16 = 73728B)
//               phase 3   -> per-wave P (1280B), ctx A-stage (2304B), out stage (4352B)
#define PS_OFF  0
#define CB_OFF  (NWAVE * 16 * KS_STRIDE * 2)                 // 20480
#define OB_OFF  (CB_OFF + NWAVE * 16 * WS_STRIDE * 2)        // 57344
#define U_BYTES (OB_OFF + NWAVE * 16 * 68 * 4)               // 126976 (>= 73728)

__global__ __launch_bounds__(512) void psab_fused(
    const float* __restrict__ x,
    const float* __restrict__ wk, const float* __restrict__ bk,
    const float* __restrict__ wv, const float* __restrict__ bv,
    const float* __restrict__ wo, const float* __restrict__ bo,
    float* __restrict__ out) {
  __shared__ __align__(16) unsigned char U[U_BYTES];     // 124.0 KB (aliased)
  __shared__ __align__(16) half_t Ks[512 * KS_STRIDE];   // 40.0 KB [tok][ckey]
  __shared__ __align__(16) half_t VTs[64 * VT_STRIDE];   // 65.0 KB [cval][tok]
  __shared__ __align__(16) half_t Wks[32 * WS_STRIDE];   // weights [o][c]
  __shared__ __align__(16) half_t Wvs[64 * WS_STRIDE];
  __shared__ __align__(16) half_t Wos[64 * WS_STRIDE];

  const int tid  = threadIdx.x;
  const int lane = tid & 31;
  const int wid  = tid >> 5;   // wave 0..15
  const int g    = lane >> 4;  // K-chunk selector
  const int nn   = lane & 15;  // row (A) / column (B/C) index

  const int win = blockIdx.x;  // 512 windows
  const int sh = win >> 6, sw = (win >> 3) & 7, sd = win & 7;
  const int winbase = sh * 8 * 4096 + sw * 8 * 64 + sd * 8;

  // ---- phase 1: stage X window + weights into LDS (f32 -> f16) ----
  half_t* Xs = (half_t*)U;  // [tok][cin], dead after phase 2
  {
    // thread <-> token (512 threads, 512 tokens); loop over 64 channels
    int hb = tid >> 6, wb = (tid >> 3) & 7, db = tid & 7;
    int sp = winbase + hb * 4096 + wb * 64 + db;
#pragma unroll 8
    for (int c = 0; c < 64; ++c)
      Xs[tid * XS_STRIDE + c] = (half_t)x[c * 262144 + sp];
  }
  for (int f = tid; f < 32 * 64; f += 512)
    Wks[(f >> 6) * WS_STRIDE + (f & 63)] = (half_t)wk[f];
  for (int f = tid; f < 64 * 64; f += 512)
    Wvs[(f >> 6) * WS_STRIDE + (f & 63)] = (half_t)wv[f];
  for (int f = tid; f < 64 * 64; f += 512)
    Wos[(f >> 6) * WS_STRIDE + (f & 63)] = (half_t)wo[f];
  __syncthreads();

  // ---- phase 2: K = X*Wk^T + bk  (Q==K),  V^T = (X*Wv^T + bv)^T ----
  for (int tt = wid; tt < 32; tt += NWAVE) {    // token tiles of 16
    const half_t* arow = Xs + (tt * 16 + nn) * XS_STRIDE;
    v16h a0 = ld_a(arow, g);        // cin 0..31
    v16h a1 = ld_a(arow + 32, g);   // cin 32..63
#pragma unroll
    for (int ct = 0; ct < 2; ++ct) {            // C_KEY = 2 tiles
      const half_t* br = Wks + (ct * 16 + nn) * WS_STRIDE;
      v8f c = {};
      c = wmma_f16(a0, ld_b(br, g), c);
      c = wmma_f16(a1, ld_b(br + 32, g), c);
      float bias = bk[ct * 16 + nn];
#pragma unroll
      for (int v = 0; v < 8; ++v)
        Ks[(tt * 16 + v + 8 * g) * KS_STRIDE + ct * 16 + nn] =
            (half_t)(c[v] + bias);
    }
#pragma unroll
    for (int ct = 0; ct < 4; ++ct) {            // C_VAL = 4 tiles
      const half_t* br = Wvs + (ct * 16 + nn) * WS_STRIDE;
      v8f c = {};
      c = wmma_f16(a0, ld_b(br, g), c);
      c = wmma_f16(a1, ld_b(br + 32, g), c);
      float bias = bv[ct * 16 + nn];
#pragma unroll
      for (int v = 0; v < 8; ++v)
        VTs[(ct * 16 + nn) * VT_STRIDE + tt * 16 + v + 8 * g] =
            (half_t)(c[v] + bias);
    }
  }
  __syncthreads();  // Xs dead from here; U region re-used per-wave

  // ---- phase 3: flash attention over 512 keys + out-projection ----
  const float sc = 0.17677669529663688f;  // 1/sqrt(32)
  half_t* Pbuf = (half_t*)(U + PS_OFF) + wid * 16 * KS_STRIDE;
  half_t* Cbuf = (half_t*)(U + CB_OFF) + wid * 16 * WS_STRIDE;
  float*  Obuf = (float*)(U + OB_OFF) + wid * 16 * 68;

#pragma unroll 1
  for (int qi = 0; qi < 2; ++qi) {
    const int qt = wid + qi * NWAVE;           // query tile (16 tokens)
    v16h qa = ld_a(Ks + (qt * 16 + nn) * KS_STRIDE, g);
    float mrow[8], lrow[8];
    v8f acc[4];
#pragma unroll
    for (int v = 0; v < 8; ++v) { mrow[v] = -1e30f; lrow[v] = 0.f; }
#pragma unroll
    for (int t = 0; t < 4; ++t) { v8f z = {}; acc[t] = z; }

    for (int kb = 0; kb < 16; ++kb) {          // 32 keys per step
      v8f z = {};
      v8f s0 = wmma_f16(qa, ld_b(Ks + (kb * 32 + nn) * KS_STRIDE, g), z);
      v8f s1 = wmma_f16(qa, ld_b(Ks + (kb * 32 + 16 + nn) * KS_STRIDE, g), z);
#pragma unroll
      for (int v = 0; v < 8; ++v) {
        float a0 = s0[v] * sc, a1 = s1[v] * sc;
        float mnew = fmaxf(mrow[v], rmax16(fmaxf(a0, a1)));
        float alpha = __expf(mrow[v] - mnew);
        float p0 = __expf(a0 - mnew);
        float p1 = __expf(a1 - mnew);
        lrow[v] = lrow[v] * alpha + rsum16(p0 + p1);
        mrow[v] = mnew;
#pragma unroll
        for (int t = 0; t < 4; ++t) acc[t][v] *= alpha;
        Pbuf[(v + 8 * g) * KS_STRIDE + nn]      = (half_t)p0;
        Pbuf[(v + 8 * g) * KS_STRIDE + 16 + nn] = (half_t)p1;
      }
      // same-wave LDS bounce: C-layout probs -> A-operand layout
      v16h pa = ld_a(Pbuf + nn * KS_STRIDE, g);
#pragma unroll
      for (int ct = 0; ct < 4; ++ct)
        acc[ct] = wmma_f16(
            pa, ld_b(VTs + (ct * 16 + nn) * VT_STRIDE + kb * 32, g), acc[ct]);
    }

#pragma unroll
    for (int v = 0; v < 8; ++v) {
      float inv = 1.0f / lrow[v];
#pragma unroll
      for (int ct = 0; ct < 4; ++ct) acc[ct][v] *= inv;
    }

    // ctx -> LDS A-stage, then out = ctx * Wo^T + bo
#pragma unroll
    for (int ct = 0; ct < 4; ++ct)
#pragma unroll
      for (int v = 0; v < 8; ++v)
        Cbuf[(v + 8 * g) * WS_STRIDE + ct * 16 + nn] = (half_t)acc[ct][v];
    v16h ca0 = ld_a(Cbuf + nn * WS_STRIDE, g);
    v16h ca1 = ld_a(Cbuf + nn * WS_STRIDE + 32, g);
#pragma unroll
    for (int ct = 0; ct < 4; ++ct) {
      const half_t* br = Wos + (ct * 16 + nn) * WS_STRIDE;
      v8f c = {};
      c = wmma_f16(ca0, ld_b(br, g), c);
      c = wmma_f16(ca1, ld_b(br + 32, g), c);
      float bias = bo[ct * 16 + nn];
#pragma unroll
      for (int v = 0; v < 8; ++v)
        Obuf[(v + 8 * g) * 68 + ct * 16 + nn] = c[v] + bias;
    }
    // d-contiguous write-back of this wave's 16 tokens x 64 channels
    for (int it = 0; it < 32; ++it) {
      int flat = it * 32 + lane;
      int c = flat >> 4, tw = flat & 15;
      int t = qt * 16 + tw;
      int hb = t >> 6, wb = (t >> 3) & 7, db = t & 7;
      out[c * 262144 + winbase + hb * 4096 + wb * 64 + db] = Obuf[tw * 68 + c];
    }
  }
}

extern "C" void kernel_launch(void* const* d_in, const int* in_sizes, int n_in,
                              void* d_out, int out_size, void* d_ws,
                              size_t ws_size, hipStream_t stream) {
  const float* x  = (const float*)d_in[0];
  const float* wk = (const float*)d_in[1];
  const float* bk = (const float*)d_in[2];
  const float* wv = (const float*)d_in[3];
  const float* bv = (const float*)d_in[4];
  const float* wo = (const float*)d_in[5];
  const float* bo = (const float*)d_in[6];
  // d_in[7] = scale (== 8), compile-time constant here
  float* out = (float*)d_out;
  psab_fused<<<dim3(512), dim3(512), 0, stream>>>(x, wk, bk, wv, bv, wo, bo,
                                                  out);
}
